// GroupedQueryAttention_53472342835626
// MI455X (gfx1250) — matrix operand order repaired and checked by reading, not compiled, not run on hardware
//
#include <hip/hip_runtime.h>

typedef __bf16 bf16;
typedef __attribute__((ext_vector_type(16))) __bf16 v16bf;
typedef __attribute__((ext_vector_type(8)))  float  v8f;
typedef unsigned int u32;
typedef unsigned long long u64;
typedef __attribute__((ext_vector_type(4))) u32 u32x4;
typedef __attribute__((ext_vector_type(8))) u32 u32x8;

#define DIMN   2048
#define SEQ    2048
#define BATCH  2
#define NQH    32
#define NKVH   8
#define HD     64
#define KVDIM  512
#define GQA    4
#define SCL    0.125f
#define NEGINF (-1.0e30f)

// ---------------------------------------------------------------------------
// RoPE table: with factor=8 and seq_eff==max_len, adj==1 -> inv_freq = theta^-idx
// ---------------------------------------------------------------------------
__global__ void rope_table_kernel(float* __restrict__ cosT, float* __restrict__ sinT) {
    int idx = blockIdx.x * 256 + threadIdx.x;
    if (idx >= SEQ * HD) return;
    int s = idx >> 6, d = idx & 63, i = d & 31;
    float invf = powf(10000.0f, -(2.0f * (float)i) / 64.0f);
    float ang = (float)s * invf;
    cosT[idx] = cosf(ang);
    sinT[idx] = sinf(ang);
}

__device__ inline float rmax16(float v) {
    v = fmaxf(v, __shfl_xor(v, 1, 32));
    v = fmaxf(v, __shfl_xor(v, 2, 32));
    v = fmaxf(v, __shfl_xor(v, 4, 32));
    v = fmaxf(v, __shfl_xor(v, 8, 32));
    return v;
}
__device__ inline float rsum16(float v) {
    v += __shfl_xor(v, 1, 32);
    v += __shfl_xor(v, 2, 32);
    v += __shfl_xor(v, 4, 32);
    v += __shfl_xor(v, 8, 32);
    return v;
}

// CDNA5 async copy: 16B/lane global -> LDS, tracked by ASYNCcnt (no VGPR transit)
__device__ inline void async_copy_b128(u32 lds_off, u64 gaddr) {
    asm volatile("global_load_async_to_lds_b128 %0, %1, off"
                 :: "v"(lds_off), "v"(gaddr) : "memory");
}
__device__ inline void wait_asynccnt0() {
    asm volatile("s_wait_asynccnt 0x0" ::: "memory");
}

// ---------------------------------------------------------------------------
// Tiled BF16 WMMA GEMM: C[128x64] per block, K-step 32, 8 waves (4Mx2N of 32x32).
// MODE 0: Q proj  (f32 A, +bias, +RoPE, bf16 out [B,QH,S,HD])
// MODE 1: K proj  (f32 A, +bias, +RoPE, bf16 out [B,KVH,S,HD])
// MODE 2: V proj  (f32 A, +bias,        bf16 out [B,KVH,S,HD])
// MODE 3: O proj  (bf16 A via ASYNC-to-LDS, +bias, f32 out [B,S,DIM])
// ---------------------------------------------------------------------------
template <int MODE>
__global__ __launch_bounds__(256) void proj_kernel(const void* __restrict__ Ain,
                                                   const float* __restrict__ W,
                                                   const float* __restrict__ bias,
                                                   const float* __restrict__ cosT,
                                                   const float* __restrict__ sinT,
                                                   void* __restrict__ Out, int Ntot) {
    __shared__ __align__(16) bf16 As[128][40];   // 128 x 32 bf16 A tile (padded)
    __shared__ __align__(16) bf16 Bs[32][72];    // 32 x 64 bf16 B tile (padded)
    __shared__ float Os[128][65];                // f32 staging for bias+RoPE epilogue

    const int tid   = threadIdx.x;
    const int mbase = blockIdx.y * 128;
    const int nbase = blockIdx.x * 64;
    const int lane  = tid & 31, wid = tid >> 5;
    const int half  = lane >> 4, lm = lane & 15;
    const int wm = (wid & 3) * 32, wn = (wid >> 2) * 32;

    v8f acc[2][2] = {};

    for (int k0 = 0; k0 < DIMN; k0 += 32) {
        __syncthreads();
        // ---- stage A tile ----
        {
            int row = tid >> 1, c0 = (tid & 1) * 16;
            if (MODE == 3) {
                // bf16 copy: bypass VGPRs with CDNA5 async LDS-DMA (2 x 16B)
                const bf16* src = (const bf16*)Ain + (size_t)(mbase + row) * DIMN + k0 + c0;
                u32 ldsa = (u32)(uintptr_t)&As[row][c0];
                u64 ga   = (u64)(uintptr_t)src;
                async_copy_b128(ldsa, ga);
                async_copy_b128(ldsa + 16u, ga + 16u);
            } else {
                const float* src = (const float*)Ain + (size_t)(mbase + row) * DIMN + k0 + c0;
#pragma unroll
                for (int i = 0; i < 16; ++i) As[row][c0 + i] = (bf16)src[i];
            }
        }
        // ---- stage B tile (f32 -> bf16 convert) ----
        {
            int r = tid >> 3, c0 = (tid & 7) * 8;
            const float* src = W + (size_t)(k0 + r) * Ntot + nbase + c0;
#pragma unroll
            for (int i = 0; i < 8; ++i) Bs[r][c0 + i] = (bf16)src[i];
        }
        if (MODE == 3) wait_asynccnt0();
        __syncthreads();

        // ---- build A/B fragments (contiguous k = 16*half + s, same order for A and B) and issue WMMA ----
        v16bf a[2], bb[2];
#pragma unroll
        for (int i = 0; i < 2; ++i) {
            int row = wm + i * 16 + lm;
#pragma unroll
            for (int j = 0; j < 8; ++j) {
                int kb = 2 * j + (half ? 16 : 0);
                a[i][2 * j]     = As[row][kb];
                a[i][2 * j + 1] = As[row][kb + 1];
            }
        }
#pragma unroll
        for (int j2 = 0; j2 < 2; ++j2) {
            int col = wn + j2 * 16 + lm;
#pragma unroll
            for (int j = 0; j < 8; ++j) {
                int kb = 2 * j + (half ? 16 : 0);
                bb[j2][2 * j]     = Bs[kb][col];
                bb[j2][2 * j + 1] = Bs[kb + 1][col];
            }
        }
#pragma unroll
        for (int i = 0; i < 2; ++i)
#pragma unroll
            for (int j2 = 0; j2 < 2; ++j2)
                acc[i][j2] = __builtin_amdgcn_wmma_f32_16x16x32_bf16(
                    false, a[i], false, bb[j2], (short)0, acc[i][j2], false, false);
    }

    float bval[2] = {bias[nbase + wn + lm], bias[nbase + wn + 16 + lm]};

    if (MODE == 3) {
        float* out = (float*)Out;
#pragma unroll
        for (int i = 0; i < 2; ++i)
#pragma unroll
            for (int j2 = 0; j2 < 2; ++j2)
#pragma unroll
                for (int r = 0; r < 8; ++r) {
                    int rowg = mbase + wm + i * 16 + r + 8 * half;
                    int colg = nbase + wn + j2 * 16 + lm;
                    out[(size_t)rowg * DIMN + colg] = acc[i][j2][r] + bval[j2];
                }
    } else {
#pragma unroll
        for (int i = 0; i < 2; ++i)
#pragma unroll
            for (int j2 = 0; j2 < 2; ++j2)
#pragma unroll
                for (int r = 0; r < 8; ++r)
                    Os[wm + i * 16 + r + 8 * half][wn + j2 * 16 + lm] = acc[i][j2][r] + bval[j2];
        __syncthreads();
        bf16* out = (bf16*)Out;
        const int NH = (MODE == 0) ? NQH : NKVH;
        const int h  = nbase >> 6;  // one head per 64-wide N tile
        for (int idx = tid; idx < 128 * 64; idx += 256) {
            int row = idx >> 6, col = idx & 63;
            int grow = mbase + row;
            int bI = grow >> 11, spos = grow & (SEQ - 1);
            float val = Os[row][col];
            if (MODE <= 1) {  // RoPE: q*cos + rotate_half(q)*sin
                float partner = Os[row][col ^ 32];
                float rot = (col < 32) ? -partner : partner;
                val = val * cosT[spos * HD + col] + rot * sinT[spos * HD + col];
            }
            out[(((size_t)bI * NH + h) * SEQ + spos) * HD + col] = (bf16)val;
        }
    }
}

// ---------------------------------------------------------------------------
// Causal GQA flash attention. K tile staged by the Tensor Data Mover (TDM):
// 2D tile 64x32 bf16, HW-padded 1 dword per 32 dwords to match Ks' 66-elem
// stride (bank-conflict padding done by the DMA engine). V staged manually
// (transposed). Online softmax; bf16 WMMA everywhere.
// ---------------------------------------------------------------------------
__global__ __launch_bounds__(128) void attn_kernel(const bf16* __restrict__ q,
                                                   const bf16* __restrict__ k,
                                                   const bf16* __restrict__ v,
                                                   bf16* __restrict__ ctx) {
    __shared__ __align__(16) bf16 Ks[32][66];      // [key][d], stride 33 dwords
    __shared__ __align__(16) bf16 Vt[64][34];      // [d][key] (transposed)
    __shared__ __align__(16) bf16 Ps[4][16][34];   // per-wave P tile 16x32

    const int tid = threadIdx.x, lane = tid & 31, wid = tid >> 5;
    const int half = lane >> 4, lm = lane & 15;
    const int bh = blockIdx.y;
    const int b = bh / NQH, h = bh % NQH, kvh = h / GQA;
    const int qbase = blockIdx.x * 64;
    const int qb = qbase + wid * 16;

    // Q fragments (A-matrix 16x64 -> two K=32 chunks), loaded once
    const bf16* qrow = q + (((size_t)b * NQH + h) * SEQ + (qb + lm)) * HD;
    v16bf aq[2];
#pragma unroll
    for (int c = 0; c < 2; ++c)
#pragma unroll
        for (int j = 0; j < 8; ++j) {
            int kb = 2 * j + (half ? 16 : 0) + 32 * c;
            aq[c][2 * j]     = qrow[kb];
            aq[c][2 * j + 1] = qrow[kb + 1];
        }

    const bf16* khead = k + ((size_t)b * NKVH + kvh) * SEQ * HD;
    const bf16* vhead = v + ((size_t)b * NKVH + kvh) * SEQ * HD;
    const u32 ksLds = (u32)(uintptr_t)&Ks[0][0];

    float mrun[8], lrun[8];
    v8f vacc[4] = {};
#pragma unroll
    for (int r = 0; r < 8; ++r) { mrun[r] = NEGINF; lrun[r] = 0.0f; }

    const int kend = qbase + 64;
    for (int k0 = 0; k0 < kend; k0 += 32) {
        __syncthreads();
        // ---- TDM: DMA K tile (32 rows x 128B, +1dw pad / 32dw) into Ks ----
        if (wid == 0) {
            u64 ga = (u64)(uintptr_t)(khead + (size_t)k0 * HD);
            u32x4 g0 = { 1u,                               // count=1, user mode
                         ksLds,                            // lds_addr
                         (u32)ga,                          // global_addr[31:0]
                         ((u32)(ga >> 32) & 0x01FFFFFFu) | 0x80000000u };  // [56:32] | type=2
            u32x8 g1 = { 0x01110000u,                      // data_size=2B, pad_en, pad_int=32dw
                         (u32)HD << 16,                    // tensor_dim0 = 64
                         (u32)SEQ << 16,                   // tensor_dim1 = 2048
                         (u32)HD << 16,                    // tile_dim0 = 64
                         32u,                              // tile_dim1 = 32
                         (u32)HD,                          // tensor_dim0_stride = 64
                         0u, 0u };
            asm volatile("tensor_load_to_lds %0, %1" :: "s"(g0), "s"(g1) : "memory");
        }
        // ---- stage V transposed (manual: DMA cannot transpose) ----
        {
            int key = tid >> 2, d0 = (tid & 3) * 16;
            const bf16* vrow = vhead + (size_t)(k0 + key) * HD + d0;
#pragma unroll
            for (int i = 0; i < 16; ++i) Vt[d0 + i][key] = vrow[i];
        }
        if (wid == 0) __builtin_amdgcn_s_wait_tensorcnt(0);
        __syncthreads();

        // ---- scores: S(16x32) = Q(16x64) @ K^T, two 16-key column tiles ----
        v8f sc[2] = {};
#pragma unroll
        for (int t2 = 0; t2 < 2; ++t2)
#pragma unroll
            for (int c = 0; c < 2; ++c) {
                v16bf bk;
                int keyr = t2 * 16 + lm;
#pragma unroll
                for (int j = 0; j < 8; ++j) {
                    int kk = 2 * j + (half ? 16 : 0) + 32 * c;
                    bk[2 * j]     = Ks[keyr][kk];
                    bk[2 * j + 1] = Ks[keyr][kk + 1];
                }
                sc[t2] = __builtin_amdgcn_wmma_f32_16x16x32_bf16(
                    false, aq[c], false, bk, (short)0, sc[t2], false, false);
            }

        // ---- online softmax (row reductions across 16-lane groups) ----
#pragma unroll
        for (int r = 0; r < 8; ++r) {
            int rowg = qb + r + 8 * half;
            float s0 = sc[0][r] * SCL; if (k0 + lm > rowg)      s0 = NEGINF;
            float s1 = sc[1][r] * SCL; if (k0 + 16 + lm > rowg) s1 = NEGINF;
            float rowm = rmax16(fmaxf(s0, s1));
            float mn = fmaxf(mrun[r], rowm);
            float fc = __expf(mrun[r] - mn);
            mrun[r] = mn;
            float p0 = __expf(s0 - mn), p1 = __expf(s1 - mn);
            lrun[r] = lrun[r] * fc + rsum16(p0 + p1);
#pragma unroll
            for (int nt = 0; nt < 4; ++nt) vacc[nt][r] *= fc;
            Ps[wid][r + 8 * half][lm]      = (bf16)p0;
            Ps[wid][r + 8 * half][16 + lm] = (bf16)p1;
        }
        __syncthreads();

        // ---- ctx += P(16x32) @ V(32x64) ----
        v16bf ap;
#pragma unroll
        for (int j = 0; j < 8; ++j) {
            int kb = 2 * j + (half ? 16 : 0);
            ap[2 * j]     = Ps[wid][lm][kb];
            ap[2 * j + 1] = Ps[wid][lm][kb + 1];
        }
#pragma unroll
        for (int nt = 0; nt < 4; ++nt) {
            v16bf bv;
            int d = nt * 16 + lm;
#pragma unroll
            for (int j = 0; j < 8; ++j) {
                int kk = 2 * j + (half ? 16 : 0);
                bv[2 * j]     = Vt[d][kk];
                bv[2 * j + 1] = Vt[d][kk + 1];
            }
            vacc[nt] = __builtin_amdgcn_wmma_f32_16x16x32_bf16(
                false, ap, false, bv, (short)0, vacc[nt], false, false);
        }
    }

    // ---- normalize and store ctx as [B,S,DIM] bf16 (col = h*HD + d) ----
#pragma unroll
    for (int nt = 0; nt < 4; ++nt)
#pragma unroll
        for (int r = 0; r < 8; ++r) {
            int spos = qb + r + 8 * half;
            int col = h * HD + nt * 16 + lm;
            float val = vacc[nt][r] / lrun[r];
            ctx[((size_t)b * SEQ + spos) * DIMN + col] = (bf16)val;
        }
}

// ---------------------------------------------------------------------------
extern "C" void kernel_launch(void* const* d_in, const int* in_sizes, int n_in,
                              void* d_out, int out_size, void* d_ws, size_t ws_size,
                              hipStream_t stream) {
    (void)in_sizes; (void)n_in; (void)out_size; (void)ws_size;
    const float* query = (const float*)d_in[0];
    const float* keyi  = (const float*)d_in[1];
    const float* value = (const float*)d_in[2];
    const float* w_q   = (const float*)d_in[3];
    const float* b_q   = (const float*)d_in[4];
    const float* w_k   = (const float*)d_in[5];
    const float* b_k   = (const float*)d_in[6];
    const float* w_v   = (const float*)d_in[7];
    const float* b_v   = (const float*)d_in[8];
    const float* w_o   = (const float*)d_in[9];
    const float* b_o   = (const float*)d_in[10];

    char* ws = (char*)d_ws;
    float* cosT = (float*)(ws);                       // 512 KB
    float* sinT = (float*)(ws + 524288);              // 512 KB
    bf16* qbf   = (bf16*)(ws + 1048576);              // 16 MB  [B,QH,S,HD]
    bf16* kbf   = (bf16*)(ws + 17825792);             //  4 MB  [B,KVH,S,HD]
    bf16* vbf   = (bf16*)(ws + 22020096);             //  4 MB  [B,KVH,S,HD]
    bf16* ctxbf = (bf16*)(ws + 26214400);             // 16 MB  [B,S,DIM]

    rope_table_kernel<<<(SEQ * HD) / 256, 256, 0, stream>>>(cosT, sinT);

    // Q / K / V projections (fused bias + RoPE + bf16 head-major store)
    proj_kernel<0><<<dim3(DIMN / 64, (BATCH * SEQ) / 128), 256, 0, stream>>>(
        query, w_q, b_q, cosT, sinT, qbf, DIMN);
    proj_kernel<1><<<dim3(KVDIM / 64, (BATCH * SEQ) / 128), 256, 0, stream>>>(
        keyi, w_k, b_k, cosT, sinT, kbf, KVDIM);
    proj_kernel<2><<<dim3(KVDIM / 64, (BATCH * SEQ) / 128), 256, 0, stream>>>(
        value, w_v, b_v, cosT, sinT, vbf, KVDIM);

    // Causal GQA flash attention (TDM-staged K tiles)
    attn_kernel<<<dim3(SEQ / 64, BATCH * NQH), 128, 0, stream>>>(qbf, kbf, vbf, ctxbf);

    // Output projection -> f32 d_out (ASYNC-to-LDS staged A tiles)
    proj_kernel<3><<<dim3(DIMN / 64, (BATCH * SEQ) / 128), 256, 0, stream>>>(
        ctxbf, w_o, b_o, cosT, sinT, (float*)d_out, DIMN);
}
